// RelPartialLearnableDecoderLayer_51754355917029
// MI455X (gfx1250) — compile-verified, matrix operand order checked by reading, not run on hardware
//
#include <hip/hip_runtime.h>
#include <hip/hip_bf16.h>

typedef unsigned short u16;
typedef unsigned int u32;
typedef __attribute__((ext_vector_type(16))) __bf16 v16bf;
typedef __attribute__((ext_vector_type(8))) float v8f;

#define QLEN 1024
#define MLEN 1024
#define KLEN 2048
#define BSZ 4
#define NH 16
#define DH 64
#define DM 1024
#define DI 4096
#define NHDH (NH * DH) // 1024

// ---------------------------------------------------------------- helpers

__device__ __forceinline__ u16 f2bf(float f) {
  u32 u = __float_as_uint(f);
  u32 r = (u + 0x7FFFu + ((u >> 16) & 1u)) >> 16;
  return (u16)r;
}

union Frag {
  v16bf v;
  u32 u[8];
};

// Load a 16-lane-row x 32-K bf16 fragment. `row_base_u` points at the start
// (in uints) of this lane's row; kbase_u selects the 32-element K chunk.
// Layout per CDNA5 ISA 7.12.2: lanes 0-15 hold K {0..7,16..23}, lanes 16-31
// hold K {8..15,24..31} of the same 16 rows/cols.
__device__ __forceinline__ v16bf load_frag(const u32* row_base_u, int kbase_u,
                                           int laneHalf) {
  Frag f;
  const u32* p = row_base_u + kbase_u + laneHalf * 4;
#pragma unroll
  for (int g = 0; g < 8; ++g) f.u[g] = p[(g & 3) + ((g >> 2) << 3)];
  return f.v;
}

__device__ __forceinline__ v8f wmma_bf16(v16bf a, v16bf b, v8f c) {
  return __builtin_amdgcn_wmma_f32_16x16x32_bf16(false, a, false, b, (short)0,
                                                 c, false, false);
}

__device__ __forceinline__ v8f vzero() {
  v8f z = {0.f, 0.f, 0.f, 0.f, 0.f, 0.f, 0.f, 0.f};
  return z;
}

// CDNA5 async DMA: global -> LDS, 16 bytes per lane, tracked by ASYNCcnt.
__device__ __forceinline__ void async_copy16(u32 lds_byte_addr,
                                             const void* gptr) {
  asm volatile("global_load_async_to_lds_b128 %0, %1, off"
               :
               : "v"(lds_byte_addr), "v"((unsigned long long)(uintptr_t)gptr)
               : "memory");
}

__device__ __forceinline__ void wait_async0() {
  asm volatile("s_wait_asynccnt 0x0" ::: "memory");
}

// ---------------------------------------------------------------- converts

__global__ __launch_bounds__(256) void conv_kernel(const float* __restrict__ s,
                                                   u16* __restrict__ d,
                                                   int count) {
  int i = blockIdx.x * 256 + threadIdx.x;
  if (i < count) d[i] = f2bf(s[i]);
}

// ---------------------------------------------------------------- generic GEMM
// C[M,N] = A[M,K] * B[N,K]^T  (A,B bf16 row-major), M,N mult of 128, K of 32.
// Compile-time epilogue: +bias[n], relu, +resid[m,n]; fp32 and/or bf16 out.
// Tiles staged via async global->LDS DMA, double buffered, 1 barrier/K-step.
template <bool HB, bool HR, bool RELU, bool OF, bool OH>
__global__ __launch_bounds__(256) void gemm_kernel(
    const u16* __restrict__ A, const u16* __restrict__ B, int M, int N, int K,
    const float* __restrict__ bias, const float* __restrict__ resid,
    float* __restrict__ outF, u16* __restrict__ outH) {
  __shared__ __align__(16) u32 As[2][128 * 20]; // 32 bf16 = 16 uints + pad
  __shared__ __align__(16) u32 Bs[2][128 * 20];

  const int tid = threadIdx.x;
  const int lane = tid & 31;
  const int wid = tid >> 5;
  const int waveM = wid >> 2; // 0..1 -> 64 rows each
  const int waveN = wid & 3;  // 0..3 -> 32 cols each
  const int laneHalf = lane >> 4;
  const int lm = lane & 15;
  const int m0 = blockIdx.y * 128;
  const int n0 = blockIdx.x * 128;
  const int Ku = K >> 1; // uints per row

  const u32* Ag = (const u32*)A;
  const u32* Bg = (const u32*)B;

  const int ldRow = tid >> 1;      // 0..127
  const int ldCol = (tid & 1) * 8; // uint col 0 or 8

  const u32* gA = Ag + (size_t)(m0 + ldRow) * Ku + ldCol;
  const u32* gB = Bg + (size_t)(n0 + ldRow) * Ku + ldCol;
  const u32 lA0 = (u32)(uintptr_t)(&As[0][0] + ldRow * 20 + ldCol);
  const u32 lB0 = (u32)(uintptr_t)(&Bs[0][0] + ldRow * 20 + ldCol);
  const u32 lbufsz = (u32)(128 * 20 * 4);

  v8f acc[4][2];
#pragma unroll
  for (int i = 0; i < 4; ++i)
#pragma unroll
    for (int j = 0; j < 2; ++j) acc[i][j] = vzero();

  const int ksteps = K >> 5;

  // prefetch tile 0 into buffer 0
  async_copy16(lA0, gA);
  async_copy16(lA0 + 16, gA + 4);
  async_copy16(lB0, gB);
  async_copy16(lB0 + 16, gB + 4);

  for (int kt = 0; kt < ksteps; ++kt) {
    wait_async0();   // this wave's pending tile writes have landed
    __syncthreads(); // all waves' writes visible; prev buffer free

    if (kt + 1 < ksteps) {
      const u32 nb = (u32)((kt + 1) & 1) * lbufsz;
      const u32* pa = gA + (size_t)(kt + 1) * 16;
      const u32* pb = gB + (size_t)(kt + 1) * 16;
      async_copy16(lA0 + nb, pa);
      async_copy16(lA0 + nb + 16, pa + 4);
      async_copy16(lB0 + nb, pb);
      async_copy16(lB0 + nb + 16, pb + 4);
    }

    const u32* Ab = As[kt & 1];
    const u32* Bb = Bs[kt & 1];
    v16bf af[4], bf[2];
#pragma unroll
    for (int fm = 0; fm < 4; ++fm)
      af[fm] = load_frag(Ab + (waveM * 64 + fm * 16 + lm) * 20, 0, laneHalf);
#pragma unroll
    for (int fn = 0; fn < 2; ++fn)
      bf[fn] = load_frag(Bb + (waveN * 32 + fn * 16 + lm) * 20, 0, laneHalf);
#pragma unroll
    for (int fm = 0; fm < 4; ++fm)
#pragma unroll
      for (int fn = 0; fn < 2; ++fn)
        acc[fm][fn] = wmma_bf16(af[fm], bf[fn], acc[fm][fn]);
  }

  // straight-line epilogue (compile-time specialized)
  const int ncol0 = n0 + waveN * 32 + lm;
  float bv0 = 0.f, bv1 = 0.f;
  if constexpr (HB) {
    bv0 = bias[ncol0];
    bv1 = bias[ncol0 + 16];
  }
#pragma unroll
  for (int fm = 0; fm < 4; ++fm) {
#pragma unroll
    for (int g = 0; g < 8; ++g) {
      const int m = m0 + waveM * 64 + fm * 16 + laneHalf * 8 + g;
      const size_t rowOff = (size_t)m * N;
      float v0 = acc[fm][0][g] + bv0;
      float v1 = acc[fm][1][g] + bv1;
      if constexpr (RELU) {
        v0 = fmaxf(v0, 0.0f);
        v1 = fmaxf(v1, 0.0f);
      }
      if constexpr (HR) {
        v0 += resid[rowOff + ncol0];
        v1 += resid[rowOff + ncol0 + 16];
      }
      if constexpr (OF) {
        outF[rowOff + ncol0] = v0;
        outF[rowOff + ncol0 + 16] = v1;
      }
      if constexpr (OH) {
        outH[rowOff + ncol0] = f2bf(v0);
        outH[rowOff + ncol0 + 16] = f2bf(v1);
      }
    }
  }
  (void)bias;
  (void)resid;
  (void)outF;
  (void)outH;
}

// ---------------------------------------------------------------- repacks

// qrw/qrr: [b][n][i][d] = bf16(w_heads_q[(MLEN+i)*4+b, n*64+d] + bias[n,d])
__global__ __launch_bounds__(256) void pack_q_kernel(
    const float* __restrict__ wh, const float* __restrict__ rwb,
    const float* __restrict__ rrb, u16* __restrict__ qrw,
    u16* __restrict__ qrr) {
  int t = blockIdx.x * 256 + threadIdx.x; // 2^22 total
  int d = t & 63;
  int i = (t >> 6) & 1023;
  int n = (t >> 16) & 15;
  int b = t >> 20;
  float v = wh[(size_t)((MLEN + i) * BSZ + b) * (3 * NHDH) + n * 64 + d];
  qrw[t] = f2bf(v + rwb[n * 64 + d]);
  qrr[t] = f2bf(v + rrb[n * 64 + d]);
}

// k: [b][n][j][d]
__global__ __launch_bounds__(256) void pack_k_kernel(const float* __restrict__ wh,
                                                     u16* __restrict__ kk) {
  int t = blockIdx.x * 256 + threadIdx.x; // 2^23 total
  int d = t & 63;
  int j = (t >> 6) & 2047;
  int n = (t >> 17) & 15;
  int b = t >> 21;
  kk[t] = f2bf(wh[(size_t)(j * BSZ + b) * (3 * NHDH) + NHDH + n * 64 + d]);
}

// vT: [b][n][d][j]
__global__ __launch_bounds__(256) void pack_v_kernel(const float* __restrict__ wh,
                                                     u16* __restrict__ vT) {
  int t = blockIdx.x * 256 + threadIdx.x; // 2^23 total
  int j = t & 2047;
  int d = (t >> 11) & 63;
  int n = (t >> 17) & 15;
  int b = t >> 21;
  vT[t] = f2bf(wh[(size_t)(j * BSZ + b) * (3 * NHDH) + 2 * NHDH + n * 64 + d]);
}

// rk: [n][j][d]
__global__ __launch_bounds__(256) void pack_rk_kernel(
    const float* __restrict__ rhk, u16* __restrict__ rk) {
  int t = blockIdx.x * 256 + threadIdx.x; // 2^21 total
  int d = t & 63;
  int j = (t >> 6) & 2047;
  int n = t >> 17;
  rk[t] = f2bf(rhk[(size_t)j * NHDH + n * 64 + d]);
}

// ---------------------------------------------------------------- fused attention
// One block = 16 query rows of one (b,n). 4 waves / 128 threads.
// Dynamic LDS: AC[16][2048] f32 + BD[16][2048] f32 (P bf16 overlays BD).
__global__ __launch_bounds__(128) void attn_kernel(
    const u16* __restrict__ qrw, const u16* __restrict__ qrr,
    const u16* __restrict__ kk, const u16* __restrict__ rk,
    const u16* __restrict__ vT, u16* __restrict__ attn_vec) {
  extern __shared__ float smem[];
  float* AC = smem;             // 16*2048 f32
  float* BD = smem + 16 * 2048; // 16*2048 f32
  u16* P = (u16*)BD;            // reuse as 16*2048 bf16

  const int tid = threadIdx.x;
  const int lane = tid & 31;
  const int wid = tid >> 5; // 0..3
  const int laneHalf = lane >> 4;
  const int lm = lane & 15;
  const int i0 = blockIdx.x * 16;
  const int bn = blockIdx.y;
  const int b = bn >> 4, n = bn & 15;

  const u32* qrw_u = (const u32*)(qrw + (size_t)bn * QLEN * DH);
  const u32* qrr_u = (const u32*)(qrr + (size_t)bn * QLEN * DH);
  const u32* kk_u = (const u32*)(kk + (size_t)bn * KLEN * DH);
  const u32* rk_u = (const u32*)(rk + (size_t)n * KLEN * DH);
  const u32* vT_u = (const u32*)(vT + (size_t)bn * DH * KLEN);

  // Q fragments for this row-tile (row stride = 64 bf16 = 32 uints)
  v16bf qwf[2], qrf[2];
#pragma unroll
  for (int kq = 0; kq < 2; ++kq) {
    qwf[kq] = load_frag(qrw_u + (size_t)(i0 + lm) * 32, kq * 16, laneHalf);
    qrf[kq] = load_frag(qrr_u + (size_t)(i0 + lm) * 32, kq * 16, laneHalf);
  }

  // Phase 1: AC = Qrw*K^T, BDraw = Qrr*R^T. Wave w owns key cols [w*512,+256)
  for (int t = 0; t < 16; ++t) {
    int j0 = wid * 512 + t * 16;
    v8f accA = vzero(), accB = vzero();
#pragma unroll
    for (int kq = 0; kq < 2; ++kq) {
      v16bf kf = load_frag(kk_u + (size_t)(j0 + lm) * 32, kq * 16, laneHalf);
      accA = wmma_bf16(qwf[kq], kf, accA);
      v16bf rf = load_frag(rk_u + (size_t)(j0 + lm) * 32, kq * 16, laneHalf);
      accB = wmma_bf16(qrf[kq], rf, accB);
    }
#pragma unroll
    for (int g = 0; g < 8; ++g) {
      int r = laneHalf * 8 + g;
      AC[r * 2048 + j0 + lm] = accA[g];
      BD[r * 2048 + j0 + lm] = accB[g];
    }
  }
  __syncthreads();

  // Phase 2: rel-shift + mask + scale -> scores in AC
  for (int e = tid; e < 16 * 2048; e += 128) {
    int r = e >> 11, j = e & 2047;
    int i = i0 + r;
    float s;
    if (j > i + MLEN) {
      s = -__builtin_inff();
    } else {
      int jr = j - i + (QLEN - 1); // exact rel_shift on unmasked region
      float bd = (jr >= 0 && jr < KLEN) ? BD[r * 2048 + jr] : 0.0f;
      s = (AC[e] + bd) * 0.125f; // 1/sqrt(64)
    }
    AC[e] = s;
  }
  __syncthreads();

  // Phase 3: softmax, write P (bf16) over BD. Wave w owns rows w*4..w*4+3.
  for (int rr = 0; rr < 4; ++rr) {
    int r = wid * 4 + rr;
    const float* row = AC + r * 2048;
    float mx = -__builtin_inff();
    for (int j = lane; j < 2048; j += 32) mx = fmaxf(mx, row[j]);
#pragma unroll
    for (int s = 16; s > 0; s >>= 1) mx = fmaxf(mx, __shfl_xor(mx, s, 32));
    float sum = 0.0f;
    for (int j = lane; j < 2048; j += 32) sum += __expf(row[j] - mx);
#pragma unroll
    for (int s = 16; s > 0; s >>= 1) sum += __shfl_xor(sum, s, 32);
    float inv = 1.0f / sum;
    for (int j = lane; j < 2048; j += 32)
      P[r * 2048 + j] = f2bf(__expf(row[j] - mx) * inv);
  }
  __syncthreads();

  // Phase 4: O[16,64] = P[16,2048] * V[2048,64]. Wave w owns d-tile w*16.
  const u32* P_u = (const u32*)P;
  const int d0 = wid * 16;
  v8f acc = vzero();
  for (int kt = 0; kt < 64; ++kt) {
    v16bf pf = load_frag(P_u + (size_t)lm * 1024, kt * 16, laneHalf);
    v16bf vf = load_frag(vT_u + (size_t)(d0 + lm) * 1024, kt * 16, laneHalf);
    acc = wmma_bf16(pf, vf, acc);
  }
#pragma unroll
  for (int g = 0; g < 8; ++g) {
    int m = laneHalf * 8 + g;
    int d = d0 + lm;
    size_t row = (size_t)((i0 + m) * BSZ + b);
    attn_vec[row * NHDH + n * DH + d] = f2bf(acc[g]);
  }
}

// ---------------------------------------------------------------- layernorm

__global__ __launch_bounds__(256) void ln_kernel(
    const float* __restrict__ x, const float* __restrict__ gamma,
    const float* __restrict__ beta, float* __restrict__ outF,
    u16* __restrict__ outH, int C) {
  __shared__ float red[256];
  const int row = blockIdx.x;
  const int tid = threadIdx.x;
  const float* xr = x + (size_t)row * C;
  float s = 0.f, s2 = 0.f;
  for (int c = tid; c < C; c += 256) {
    float v = xr[c];
    s += v;
    s2 += v * v;
  }
  red[tid] = s;
  __syncthreads();
  for (int o = 128; o > 0; o >>= 1) {
    if (tid < o) red[tid] += red[tid + o];
    __syncthreads();
  }
  float mean = red[0] / C;
  __syncthreads();
  red[tid] = s2;
  __syncthreads();
  for (int o = 128; o > 0; o >>= 1) {
    if (tid < o) red[tid] += red[tid + o];
    __syncthreads();
  }
  float var = red[0] / C - mean * mean;
  float rstd = rsqrtf(var + 1e-5f);
  for (int c = tid; c < C; c += 256) {
    float y = (xr[c] - mean) * rstd * gamma[c] + beta[c];
    if (outF) outF[(size_t)row * C + c] = y;
    if (outH) outH[(size_t)row * C + c] = f2bf(y);
  }
}

// ---------------------------------------------------------------- launch

extern "C" void kernel_launch(void* const* d_in, const int* in_sizes, int n_in,
                              void* d_out, int out_size, void* d_ws,
                              size_t ws_size, hipStream_t stream) {
  const float* w = (const float*)d_in[0];
  const float* r = (const float*)d_in[1];
  const float* r_w_bias = (const float*)d_in[2];
  const float* r_r_bias = (const float*)d_in[3];
  const float* mems = (const float*)d_in[4];
  const float* qkv_w = (const float*)d_in[5];
  const float* r_w = (const float*)d_in[6];
  const float* o_w = (const float*)d_in[7];
  const float* ln1_g = (const float*)d_in[8];
  const float* ln1_b = (const float*)d_in[9];
  const float* ln2_g = (const float*)d_in[10];
  const float* ln2_b = (const float*)d_in[11];
  const float* ff_w0 = (const float*)d_in[12];
  const float* ff_b0 = (const float*)d_in[13];
  const float* ff_w1 = (const float*)d_in[14];
  const float* ff_b1 = (const float*)d_in[15];
  // d_in[16] = attn_mask (bool) — mask is recomputed analytically in-kernel.

  char* p = (char*)d_ws;
  auto carve = [&](size_t bytes) {
    char* q = p;
    p += (bytes + 255) & ~(size_t)255;
    return q;
  };
  const int MKB = KLEN * BSZ; // 8192
  const int MQB = QLEN * BSZ; // 4096
  float* w_heads = (float*)carve((size_t)MKB * 3 * NHDH * 4); // 96 MB
  float* r_head_k = (float*)carve((size_t)KLEN * NHDH * 4);   // 8 MB
  u16* cat_h = (u16*)carve((size_t)MKB * DM * 2);             // 16 MB
  u16* qkv_h = (u16*)carve((size_t)3 * NHDH * DM * 2);        // 6 MB
  u16* r_h = (u16*)carve((size_t)KLEN * DM * 2);              // 4 MB
  u16* rw_h = (u16*)carve((size_t)NHDH * DM * 2);             // 2 MB
  u16* ow_h = (u16*)carve((size_t)DM * NHDH * 2);             // 2 MB
  u16* ffw0_h = (u16*)carve((size_t)DI * DM * 2);             // 8 MB
  u16* ffw1_h = (u16*)carve((size_t)DM * DI * 2);             // 8 MB
  u16* qrw_h = (u16*)carve((size_t)BSZ * NH * QLEN * DH * 2); // 8 MB
  u16* qrr_h = (u16*)carve((size_t)BSZ * NH * QLEN * DH * 2); // 8 MB
  u16* k_h = (u16*)carve((size_t)BSZ * NH * KLEN * DH * 2);   // 16 MB
  u16* vT_h = (u16*)carve((size_t)BSZ * NH * DH * KLEN * 2);  // 16 MB
  u16* rk_h = (u16*)carve((size_t)NH * KLEN * DH * 2);        // 4 MB
  u16* avec_h = (u16*)carve((size_t)MQB * NHDH * 2);          // 8 MB
  float* res1 = (float*)carve((size_t)MQB * DM * 4);          // 16 MB
  float* out1 = (float*)carve((size_t)MQB * DM * 4);          // 16 MB
  u16* out1_h = (u16*)carve((size_t)MQB * DM * 2);            // 8 MB
  u16* hmid_h = (u16*)carve((size_t)MQB * DI * 2);            // 32 MB
  float* res2 = (float*)carve((size_t)MQB * DM * 4);          // 16 MB

  auto cgrid = [](int count) { return dim3((count + 255) / 256); };

  // 1) fp32 -> bf16 converts (cat = [mems; w])
  conv_kernel<<<cgrid(MLEN * BSZ * DM), 256, 0, stream>>>(mems, cat_h,
                                                          MLEN * BSZ * DM);
  conv_kernel<<<cgrid(QLEN * BSZ * DM), 256, 0, stream>>>(
      w, cat_h + (size_t)MLEN * BSZ * DM, QLEN * BSZ * DM);
  conv_kernel<<<cgrid(3 * NHDH * DM), 256, 0, stream>>>(qkv_w, qkv_h,
                                                        3 * NHDH * DM);
  conv_kernel<<<cgrid(KLEN * DM), 256, 0, stream>>>(r, r_h, KLEN * DM);
  conv_kernel<<<cgrid(NHDH * DM), 256, 0, stream>>>(r_w, rw_h, NHDH * DM);
  conv_kernel<<<cgrid(DM * NHDH), 256, 0, stream>>>(o_w, ow_h, DM * NHDH);
  conv_kernel<<<cgrid(DI * DM), 256, 0, stream>>>(ff_w0, ffw0_h, DI * DM);
  conv_kernel<<<cgrid(DM * DI), 256, 0, stream>>>(ff_w1, ffw1_h, DM * DI);

  // 2) w_heads = cat @ qkv_w^T   [8192 x 3072]
  gemm_kernel<false, false, false, true, false>
      <<<dim3((3 * NHDH) / 128, MKB / 128), 256, 0, stream>>>(
          cat_h, qkv_h, MKB, 3 * NHDH, DM, nullptr, nullptr, w_heads, nullptr);
  // 3) r_head_k = r @ r_w^T      [2048 x 1024]
  gemm_kernel<false, false, false, true, false>
      <<<dim3(NHDH / 128, KLEN / 128), 256, 0, stream>>>(
          r_h, rw_h, KLEN, NHDH, DM, nullptr, nullptr, r_head_k, nullptr);

  // 4) repack per-head bf16 tensors (bias add fused into q)
  pack_q_kernel<<<cgrid(BSZ * NH * QLEN * DH), 256, 0, stream>>>(
      w_heads, r_w_bias, r_r_bias, qrw_h, qrr_h);
  pack_k_kernel<<<cgrid(BSZ * NH * KLEN * DH), 256, 0, stream>>>(w_heads, k_h);
  pack_v_kernel<<<cgrid(BSZ * NH * KLEN * DH), 256, 0, stream>>>(w_heads, vT_h);
  pack_rk_kernel<<<cgrid(NH * KLEN * DH), 256, 0, stream>>>(r_head_k, rk_h);

  // 5) fused attention: AC+BD GEMMs, rel-shift, mask, softmax, P*V
  attn_kernel<<<dim3(QLEN / 16, BSZ * NH), 128, 2 * 16 * 2048 * sizeof(float),
                stream>>>(qrw_h, qrr_h, k_h, rk_h, vT_h, avec_h);

  // 6) res1 = w + attn_vec @ o_w^T
  gemm_kernel<false, true, false, true, false>
      <<<dim3(DM / 128, MQB / 128), 256, 0, stream>>>(
          avec_h, ow_h, MQB, DM, NHDH, nullptr, w, res1, nullptr);
  // 7) out1 = LN(res1)
  ln_kernel<<<dim3(MQB), 256, 0, stream>>>(res1, ln1_g, ln1_b, out1, out1_h,
                                           DM);
  // 8) h = relu(out1 @ ff_w0^T + ff_b0)  (bf16 out)
  gemm_kernel<true, false, true, false, true>
      <<<dim3(DI / 128, MQB / 128), 256, 0, stream>>>(
          out1_h, ffw0_h, MQB, DI, DM, ff_b0, nullptr, nullptr, hmid_h);
  // 9) res2 = out1 + h @ ff_w1^T + ff_b1
  gemm_kernel<true, true, false, true, false>
      <<<dim3(DM / 128, MQB / 128), 256, 0, stream>>>(
          hmid_h, ffw1_h, MQB, DM, DI, ff_b1, out1, res2, nullptr);
  // 10) out2 = LN(res2) -> d_out
  ln_kernel<<<dim3(MQB), 256, 0, stream>>>(res2, ln2_g, ln2_b, (float*)d_out,
                                           nullptr, DM);

  (void)in_sizes;
  (void)n_in;
  (void)out_size;
  (void)ws_size;
}
